// MoE_24000277250502
// MI455X (gfx1250) — compile-verified, matrix operand order checked by reading, not run on hardware
//
#include <hip/hip_runtime.h>

#define D_DIM 1024
#define H_DIM 2048
#define E_NUM 8
#define N_TOK 8192
#define NK    (N_TOK * 2)
#define MT    16                 // token tile (one WMMA row tile)
#define XLD   (D_DIM + 8)        // padded LDS strides (bank-conflict avoidance)
#define HLD   (H_DIM + 8)

typedef __attribute__((ext_vector_type(16))) __bf16 v16bf;
typedef __attribute__((ext_vector_type(8)))  float  v8f;

// ---- workspace layout ----
// ints: [0..7]=counts  [8..15]=bucket base  [16..23]=scatter cursor  [24..32]=tile prefix
#define WS_COUNTS 0
#define WS_BASE   8
#define WS_CUR    16
#define WS_TILE   24
#define WS_HDR_BYTES 256
// small region: header + buckets + gating = 262,400 B; round to 262,656 for alignment
#define WS_SWZ_OFF 262656ull
#define W1_ELEMS  ((size_t)E_NUM * D_DIM * H_DIM)   // 16.78M
#define W2_ELEMS  ((size_t)E_NUM * H_DIM * D_DIM)   // 16.78M
// planes: W1h, W1l, W2h, W2l (bf16)
#define WS_NEEDED (WS_SWZ_OFF + 2 * (W1_ELEMS + W2_ELEMS) * 2ull)

// ------------------------------------------------------------------
// Kernel 1: gating (exact fp32 — top-k is a discrete decision)
// ------------------------------------------------------------------
__global__ __launch_bounds__(256) void moe_gating(
    const float* __restrict__ x, const float* __restrict__ Wg,
    const float* __restrict__ bg, const float* __restrict__ noise,
    int* __restrict__ hdr, int* __restrict__ ge, float* __restrict__ gw) {
  __shared__ float lg[256];
  const int tid = threadIdx.x;
  const int tok = blockIdx.x * 32 + (tid >> 3);
  const int e   = tid & 7;
  const float* xr = x + (size_t)tok * D_DIM;
  float acc = bg[e];
#pragma unroll 8
  for (int d = 0; d < D_DIM; ++d)
    acc = fmaf(xr[d], Wg[d * E_NUM + e], acc);
  acc += 0.1f * noise[(size_t)tok * E_NUM + e];
  lg[tid] = acc;
  __syncthreads();
  if (tid < 32) {
    const float* row = lg + tid * 8;
    float v0 = -1e30f, v1 = -1e30f;
    int i0 = 0, i1 = 0;
#pragma unroll
    for (int j = 0; j < E_NUM; ++j) {
      float v = row[j];
      if (v > v0)      { v1 = v0; i1 = i0; v0 = v; i0 = j; }
      else if (v > v1) { v1 = v;  i1 = j; }
    }
    float ex = expf(v1 - v0);
    float w0 = 1.0f / (1.0f + ex);
    float w1 = ex * w0;
    int tk = blockIdx.x * 32 + tid;
    ge[2 * tk]     = i0;  ge[2 * tk + 1] = i1;
    gw[2 * tk]     = w0;  gw[2 * tk + 1] = w1;
    atomicAdd(&hdr[WS_COUNTS + i0], 1);
    atomicAdd(&hdr[WS_COUNTS + i1], 1);
  }
}

// ------------------------------------------------------------------
// Kernel 2: tiny scan
// ------------------------------------------------------------------
__global__ void moe_scan(int* hdr) {
  if (threadIdx.x == 0 && blockIdx.x == 0) {
    int b = 0, tb = 0;
    hdr[WS_TILE] = 0;
    for (int e = 0; e < E_NUM; ++e) {
      hdr[WS_BASE + e] = b;
      hdr[WS_CUR + e]  = b;
      int c = hdr[WS_COUNTS + e];
      b  += c;
      tb += (c + MT - 1) / MT;
      hdr[WS_TILE + e + 1] = tb;
    }
  }
}

// ------------------------------------------------------------------
// Kernel 3: scatter tokens into per-expert buckets
// ------------------------------------------------------------------
__global__ __launch_bounds__(256) void moe_scatter(
    const int* __restrict__ ge, const float* __restrict__ gw,
    int* hdr, int* __restrict__ btok, float* __restrict__ bw) {
  int t = blockIdx.x * blockDim.x + threadIdx.x;
  if (t >= N_TOK) return;
  int e0 = ge[2 * t], e1 = ge[2 * t + 1];
  int p0 = atomicAdd(&hdr[WS_CUR + e0], 1);
  btok[p0] = t; bw[p0] = gw[2 * t];
  int p1 = atomicAdd(&hdr[WS_CUR + e1], 1);
  btok[p1] = t; bw[p1] = gw[2 * t + 1];
}

// ------------------------------------------------------------------
// WMMA fragment index helpers (CDNA5 16-bit 16x32 layout)
// ------------------------------------------------------------------
__device__ __host__ __forceinline__ int frag_k(int kg, int j) {
  int v = j >> 1;
  return ((v < 4) ? (kg + 2 * v) : (16 + kg + 2 * (v - 4))) + (j & 1);
}

__device__ __forceinline__ void load_a_frag(const __bf16* __restrict__ P, int ld,
                                            int kbase, int m, int kg, v16bf& a) {
  const __bf16* r = P + (size_t)m * ld;
#pragma unroll
  for (int v = 0; v < 8; ++v) {
    int k0 = kbase + ((v < 4) ? (kg + 2 * v) : (16 + kg + 2 * (v - 4)));
    a[2 * v]     = r[k0];
    a[2 * v + 1] = r[k0 + 1];
  }
}

// fallback: convert f32 weights in-loop (used only if ws is too small)
__device__ __forceinline__ void load_b_frag(const float* __restrict__ W, int ldn,
                                            int kbase, int n, int kg,
                                            v16bf& bh, v16bf& bl) {
#pragma unroll
  for (int v = 0; v < 8; ++v) {
    int k0 = kbase + ((v < 4) ? (kg + 2 * v) : (16 + kg + 2 * (v - 4)));
    float f0 = W[(size_t)k0 * ldn + n];
    float f1 = W[(size_t)(k0 + 1) * ldn + n];
    __bf16 h0 = (__bf16)f0, h1 = (__bf16)f1;
    bh[2 * v]     = h0;
    bh[2 * v + 1] = h1;
    bl[2 * v]     = (__bf16)(f0 - (float)h0);
    bl[2 * v + 1] = (__bf16)(f1 - (float)h1);
  }
}

// ------------------------------------------------------------------
// Kernel 0: pre-swizzle weights -> bf16 hi/lo planes in exact WMMA-B
// fragment order: dest[tile*512 + lane*16 + j].  One block per 32x16 tile.
// ------------------------------------------------------------------
__global__ __launch_bounds__(256) void moe_swizzle(
    const float* __restrict__ W, __bf16* __restrict__ dh, __bf16* __restrict__ dl,
    int Kdim, int Ndim) {
  const int kt_n = Kdim / 32;
  const int tiles_per_e = (Ndim / 16) * kt_n;
  const int b  = blockIdx.x;
  const int e  = b / tiles_per_e;
  const int r  = b % tiles_per_e;
  const int nt = r / kt_n;
  const int kt = r % kt_n;
  const float* We = W + (size_t)e * Kdim * Ndim;
  __bf16* th = dh + (size_t)b * 512;
  __bf16* tl = dl + (size_t)b * 512;
#pragma unroll
  for (int i = 0; i < 2; ++i) {
    int flat = 2 * threadIdx.x + i;
    int l = flat >> 4, j = flat & 15;
    int kg = (l >> 4) * 8;
    int k  = kt * 32 + frag_k(kg, j);
    int n  = nt * 16 + (l & 15);
    float f = We[(size_t)k * Ndim + n];
    __bf16 h = (__bf16)f;
    th[flat] = h;
    tl[flat] = (__bf16)(f - (float)h);
  }
}

// ------------------------------------------------------------------
// Kernel 4: fused grouped-expert FFN (templated on weight path).
// Block = 16 gathered tokens of one expert, 256 threads (8 waves).
// ------------------------------------------------------------------
template <bool SWZ>
__global__ __launch_bounds__(256) void moe_ffn(
    const float* __restrict__ x,
    const float* __restrict__ W1, const float* __restrict__ b1,
    const float* __restrict__ W2, const float* __restrict__ b2,
    const __bf16* __restrict__ w1h, const __bf16* __restrict__ w1l,
    const __bf16* __restrict__ w2h, const __bf16* __restrict__ w2l,
    const int* __restrict__ hdr, const int* __restrict__ btok,
    const float* __restrict__ bw, float* __restrict__ out) {
  extern __shared__ __align__(16) char smem[];
  __bf16* Xh = (__bf16*)smem;                               // [MT][XLD]
  __bf16* Xl = Xh + MT * XLD;
  __bf16* Hh = (__bf16*)(smem + (size_t)2 * MT * XLD * 2);  // [MT][HLD]
  __bf16* Hl = Hh + MT * HLD;
  __shared__ int   toks[MT];
  __shared__ float wts[MT];

  const int tb = blockIdx.x;
  const int total = hdr[WS_TILE + E_NUM];
  if (tb >= total) return;                      // block-uniform exit
  int e = 0;
  while (e < E_NUM - 1 && hdr[WS_TILE + e + 1] <= tb) ++e;
  const int tile   = tb - hdr[WS_TILE + e];
  const int bstart = hdr[WS_BASE + e];
  const int cnt    = hdr[WS_COUNTS + e];
  const int row0   = bstart + tile * MT;
  const int rows   = min(MT, cnt - tile * MT);

  const int tid = threadIdx.x;
  if (tid < MT) {
    if (tid < rows) { toks[tid] = btok[row0 + tid]; wts[tid] = bw[row0 + tid]; }
    else            { toks[tid] = -1;               wts[tid] = 0.0f; }
  }
  __syncthreads();

  // stage gathered X rows into LDS as bf16 hi/lo (zeros for padded rows)
  for (int i = tid; i < MT * (D_DIM / 4); i += 256) {
    int r = i / (D_DIM / 4);
    int c = (i % (D_DIM / 4)) * 4;
    float4 v = make_float4(0.f, 0.f, 0.f, 0.f);
    int tk = toks[r];
    if (tk >= 0) v = *(const float4*)(x + (size_t)tk * D_DIM + c);
    float f[4] = {v.x, v.y, v.z, v.w};
#pragma unroll
    for (int j = 0; j < 4; ++j) {
      __bf16 h = (__bf16)f[j];
      Xh[r * XLD + c + j] = h;
      Xl[r * XLD + c + j] = (__bf16)(f[j] - (float)h);
    }
  }
  __syncthreads();

  const int wave = tid >> 5;
  const int lane = tid & 31;
  const int n15  = lane & 15;
  const int kg   = (lane >> 4) * 8;
  const int mhi  = (lane >> 4) << 3;

  // ---------------- stage 1: h = relu(X*W1 + b1) ----------------
  const float* W1e = W1 + (size_t)e * D_DIM * H_DIM;
  const float* b1e = b1 + (size_t)e * H_DIM;
  for (int ct = wave; ct < H_DIM / 16; ct += 8) {
    const int nb = ct * 16;
    v8f acc = {0.f, 0.f, 0.f, 0.f, 0.f, 0.f, 0.f, 0.f};
    const size_t tbase = ((size_t)(e * (H_DIM / 16) + ct) * (D_DIM / 32)) * 512 + (size_t)lane * 16;
    for (int kt = 0; kt < D_DIM / 32; ++kt) {
      const int kb = kt * 32;
      v16bf ah, al, bh, bl;
      load_a_frag(Xh, XLD, kb, n15, kg, ah);
      load_a_frag(Xl, XLD, kb, n15, kg, al);
      if (SWZ) {
        const size_t off = tbase + (size_t)kt * 512;
        bh = *(const v16bf*)(w1h + off);
        bl = *(const v16bf*)(w1l + off);
      } else {
        load_b_frag(W1e, H_DIM, kb, nb + n15, kg, bh, bl);
      }
      acc = __builtin_amdgcn_wmma_f32_16x16x32_bf16(false, ah, false, bh, (short)0, acc, false, false);
      acc = __builtin_amdgcn_wmma_f32_16x16x32_bf16(false, ah, false, bl, (short)0, acc, false, false);
      acc = __builtin_amdgcn_wmma_f32_16x16x32_bf16(false, al, false, bh, (short)0, acc, false, false);
    }
    const int col = nb + n15;
    const float bias = b1e[col];
#pragma unroll
    for (int r = 0; r < 8; ++r) {
      int m = r + mhi;
      float h = acc[r] + bias;
      h = h > 0.f ? h : 0.f;
      __bf16 hh = (__bf16)h;
      Hh[m * HLD + col] = hh;
      Hl[m * HLD + col] = (__bf16)(h - (float)hh);
    }
  }
  __syncthreads();

  // ---------------- stage 2: out += w * (h*W2 + b2) ----------------
  const float* W2e = W2 + (size_t)e * H_DIM * D_DIM;
  const float* b2e = b2 + (size_t)e * D_DIM;
  for (int ct = wave; ct < D_DIM / 16; ct += 8) {
    const int nb = ct * 16;
    v8f acc = {0.f, 0.f, 0.f, 0.f, 0.f, 0.f, 0.f, 0.f};
    const size_t tbase = ((size_t)(e * (D_DIM / 16) + ct) * (H_DIM / 32)) * 512 + (size_t)lane * 16;
    for (int kt = 0; kt < H_DIM / 32; ++kt) {
      const int kb = kt * 32;
      v16bf ah, al, bh, bl;
      load_a_frag(Hh, HLD, kb, n15, kg, ah);
      load_a_frag(Hl, HLD, kb, n15, kg, al);
      if (SWZ) {
        const size_t off = tbase + (size_t)kt * 512;
        bh = *(const v16bf*)(w2h + off);
        bl = *(const v16bf*)(w2l + off);
      } else {
        load_b_frag(W2e, D_DIM, kb, nb + n15, kg, bh, bl);
      }
      acc = __builtin_amdgcn_wmma_f32_16x16x32_bf16(false, ah, false, bh, (short)0, acc, false, false);
      acc = __builtin_amdgcn_wmma_f32_16x16x32_bf16(false, ah, false, bl, (short)0, acc, false, false);
      acc = __builtin_amdgcn_wmma_f32_16x16x32_bf16(false, al, false, bh, (short)0, acc, false, false);
    }
    const int col = nb + n15;
    const float bias = b2e[col];
#pragma unroll
    for (int r = 0; r < 8; ++r) {
      int m = r + mhi;
      int tk = toks[m];
      if (tk >= 0)
        atomicAdd(out + (size_t)tk * D_DIM + col, wts[m] * (acc[r] + bias));
    }
  }
}

// ------------------------------------------------------------------
extern "C" void kernel_launch(void* const* d_in, const int* in_sizes, int n_in,
                              void* d_out, int out_size, void* d_ws, size_t ws_size,
                              hipStream_t stream) {
  const float* x     = (const float*)d_in[0];
  const float* W1    = (const float*)d_in[1];
  const float* b1    = (const float*)d_in[2];
  const float* W2    = (const float*)d_in[3];
  const float* b2    = (const float*)d_in[4];
  const float* Wg    = (const float*)d_in[5];
  const float* bg    = (const float*)d_in[6];
  const float* noise = (const float*)d_in[7];
  float* out = (float*)d_out;

  char*  ws   = (char*)d_ws;
  int*   hdr  = (int*)ws;
  int*   btok = (int*)(ws + WS_HDR_BYTES);
  float* bwts = (float*)(ws + WS_HDR_BYTES + (size_t)NK * 4);
  int*   ge   = (int*)(ws + WS_HDR_BYTES + (size_t)NK * 8);
  float* gw   = (float*)(ws + WS_HDR_BYTES + (size_t)NK * 8 + (size_t)N_TOK * 8);

  __bf16* w1h = (__bf16*)(ws + WS_SWZ_OFF);
  __bf16* w1l = w1h + W1_ELEMS;
  __bf16* w2h = w1l + W1_ELEMS;
  __bf16* w2l = w2h + W2_ELEMS;

  const bool swz = ws_size >= WS_NEEDED;   // fixed per environment -> deterministic

  hipMemsetAsync(d_out, 0, (size_t)out_size * sizeof(float), stream);
  hipMemsetAsync(d_ws, 0, WS_HDR_BYTES, stream);

  if (swz) {
    // one 32x16 weight tile per block, fragment-order bf16 hi/lo planes
    moe_swizzle<<<E_NUM * (H_DIM / 16) * (D_DIM / 32), 256, 0, stream>>>(W1, w1h, w1l, D_DIM, H_DIM);
    moe_swizzle<<<E_NUM * (D_DIM / 16) * (H_DIM / 32), 256, 0, stream>>>(W2, w2h, w2l, H_DIM, D_DIM);
  }

  moe_gating<<<N_TOK / 32, 256, 0, stream>>>(x, Wg, bg, noise, hdr, ge, gw);
  moe_scan<<<1, 1, 0, stream>>>(hdr);
  moe_scatter<<<N_TOK / 256, 256, 0, stream>>>(ge, gw, hdr, btok, bwts);

  const int maxTiles = NK / MT + E_NUM;  // 1032 (covers per-expert partial tiles)
  const size_t smem = (size_t)2 * MT * XLD * 2 + (size_t)2 * MT * HLD * 2;  // ~197.6 KB
  if (swz)
    moe_ffn<true><<<maxTiles, 256, smem, stream>>>(x, W1, b1, W2, b2,
                                                   w1h, w1l, w2h, w2l,
                                                   hdr, btok, bwts, out);
  else
    moe_ffn<false><<<maxTiles, 256, smem, stream>>>(x, W1, b1, W2, b2,
                                                    w1h, w1l, w2h, w2l,
                                                    hdr, btok, bwts, out);
}
